// ImageCaptioner_55860344651921
// MI455X (gfx1250) — matrix operand order repaired and checked by reading, not compiled, not run on hardware
//
#include <hip/hip_runtime.h>
#include <hip/hip_bf16.h>
#include <math.h>

typedef __attribute__((ext_vector_type(16))) __bf16        v16bf;
typedef __attribute__((ext_vector_type(8)))  float         v8f;
typedef __attribute__((ext_vector_type(4)))  unsigned int  v4u;
typedef __attribute__((ext_vector_type(2)))  unsigned int  v2u;

#define BN_EPS 1e-5f

// ---------------------------------------------------------------------------
// WMMA fragment loaders operating on dword-packed LDS tiles.
// A tile in LDS: dword s32[m*ldw + kp] packs (K=2kp, K=2kp+1) of row m.
// B tile in LDS (N-major): dword s32[n*ldw + kp] packs (K=2kp, K=2kp+1).
// Each lane's fragment is 2x 16B-contiguous chunks -> 2x ds_load_b128.
// ---------------------------------------------------------------------------
union FragU { v4u q[2]; v16bf v; };

__device__ __forceinline__ v16bf fragA32(const unsigned int* s32, int ldw) {
  int lane = threadIdx.x & 31;
  int m   = lane & 15;
  int kh2 = (lane >> 4) << 2;            // 0 or 4 dwords
  FragU u;
  u.q[0] = *(const v4u*)(s32 + m * ldw + kh2);
  u.q[1] = *(const v4u*)(s32 + m * ldw + 8 + kh2);
  return u.v;
}

__device__ __forceinline__ v16bf fragB32(const unsigned int* s32, int ldw) {
  int lane = threadIdx.x & 31;
  int n   = lane & 15;
  int kb2 = (lane >> 4) << 2;            // 0 or 4 dwords
  FragU u;
  u.q[0] = *(const v4u*)(s32 + n * ldw + kb2);
  u.q[1] = *(const v4u*)(s32 + n * ldw + 8 + kb2);
  return u.v;
}

// ---------------------------------------------------------------------------
// Conv3x3 (SAME) + BN + ReLU as implicit GEMM on WMMA.
// M=Cout, N=B*D*D (multiples of 64), K=Cin*9 zero-padded to Kpad (mult of 32).
// 128 threads = 4 waves; block tile 64x64; wave tile 32x32 -> 4 WMMAs/iter.
// Double-buffered LDS tiles: fetch chunk k+1 while WMMAs run on chunk k;
// one barrier per iteration.
// ---------------------------------------------------------------------------
__global__ __launch_bounds__(128)
void conv3x3_bn_relu_wmma(const __bf16* __restrict__ in, const __bf16* __restrict__ wt,
                          const float* __restrict__ gamma, const float* __restrict__ beta,
                          const float* __restrict__ mean, const float* __restrict__ var,
                          __bf16* __restrict__ out, int Cin, int Cout, int D)
{
  __shared__ unsigned int As32[2][64 * 20];   // 64 rows x 16 dwords (+4 pad)
  __shared__ unsigned int Bs32[2][64 * 20];   // 64 cols x 16 dwords (+4 pad)
  const int ldw = 20;
  const int tid  = threadIdx.x;
  const int wave = tid >> 5;
  const int wm = wave >> 1, wn = wave & 1;
  const int m0 = blockIdx.x * 64;
  const int p0 = blockIdx.y * 64;
  const int Kc   = Cin * 9;
  const int Kpad = (Kc + 31) & ~31;
  const int Kpw  = Kpad >> 1;              // dwords per weight row
  const int HW = D * D;
  const unsigned int* wt32 = (const unsigned int*)wt;
  const unsigned short* in16 = (const unsigned short*)in;

  // A staging geometry: thread covers half a row (8 dwords)
  const int ar  = tid >> 1;
  const int ac0 = (tid & 1) * 8;
  const unsigned int* arow = wt32 + (size_t)(m0 + ar) * Kpw + ac0;

  // ---- hoist per-slot pixel decomposition (invariant across K loop)
  int sY[8], sX[8];
  const unsigned short* sBase[8];
#pragma unroll
  for (int i = 0; i < 8; ++i) {
    int e = tid + i * 128;
    int n = e >> 4;
    int p = p0 + n;
    int img = p / HW, rem = p - img * HW;
    int y = rem / D, x = rem - y * D;
    sY[i] = y; sX[i] = x;
    sBase[i] = in16 + (size_t)img * Cin * HW;
  }

  // fetch helpers (registers <- global)
  v4u a0, a1;
  unsigned int bt[8];
  auto fetch = [&](int k0) {
    int k02 = k0 >> 1;
    a0 = *(const v4u*)(arow + k02);
    a1 = *(const v4u*)(arow + k02 + 4);
#pragma unroll
    for (int i = 0; i < 8; ++i) {
      int e = tid + i * 128;
      int kp = e & 15;
      unsigned int d = 0;
#pragma unroll
      for (int h = 0; h < 2; ++h) {
        int k = k0 + 2 * kp + h;
        int ci = k / 9, r = k - ci * 9;
        int kh = r / 3, kw = r - kh * 3;
        int yy = sY[i] + kh - 1, xx = sX[i] + kw - 1;
        bool valid = (k < Kc) && (yy >= 0) && (yy < D) && (xx >= 0) && (xx < D);
        size_t idx = valid ? ((size_t)ci * HW + yy * D + xx) : 0;
        unsigned int bits = sBase[i][idx];
        bits = valid ? bits : 0u;
        d |= bits << (16 * h);
      }
      bt[i] = d;
    }
  };

  v8f acc00 = {}, acc01 = {}, acc10 = {}, acc11 = {};
  const int nIter = (Kc + 31) >> 5;

  fetch(0);                               // prologue
  for (int it = 0; it < nIter; ++it) {
    unsigned int* Ab = As32[it & 1];
    unsigned int* Bb = Bs32[it & 1];
    // ---- drain registers into this iteration's LDS buffer
    *(v4u*)(&Ab[ar * ldw + ac0])     = a0;
    *(v4u*)(&Ab[ar * ldw + ac0 + 4]) = a1;
#pragma unroll
    for (int i = 0; i < 8; ++i) {
      int e = tid + i * 128;
      Bb[(e >> 4) * ldw + (e & 15)] = bt[i];
    }
    __syncthreads();
    // ---- issue next chunk's global fetch (overlaps with WMMAs below)
    if (it + 1 < nIter) fetch((it + 1) * 32);
    // ---- compute on current buffer
    v16bf fa0 = fragA32(Ab + (wm * 32) * ldw, ldw);
    v16bf fa1 = fragA32(Ab + (wm * 32 + 16) * ldw, ldw);
    v16bf fb0 = fragB32(Bb + (wn * 32) * ldw, ldw);
    v16bf fb1 = fragB32(Bb + (wn * 32 + 16) * ldw, ldw);
    acc00 = __builtin_amdgcn_wmma_f32_16x16x32_bf16(false, fa0, false, fb0, (short)0, acc00, false, false);
    acc01 = __builtin_amdgcn_wmma_f32_16x16x32_bf16(false, fa0, false, fb1, (short)0, acc01, false, false);
    acc10 = __builtin_amdgcn_wmma_f32_16x16x32_bf16(false, fa1, false, fb0, (short)0, acc10, false, false);
    acc11 = __builtin_amdgcn_wmma_f32_16x16x32_bf16(false, fa1, false, fb1, (short)0, acc11, false, false);
  }

  // ---- fused BN + ReLU epilogue (scale/shift hoisted), scatter bf16 NCHW
  int lane = tid & 31;
#pragma unroll
  for (int am = 0; am < 2; ++am) {
    int mrow = m0 + wm * 32 + am * 16 + ((lane >> 4) << 3);
    float sc[8], sh[8];
#pragma unroll
    for (int i = 0; i < 8; ++i) {
      int co = mrow + i;
      float s = gamma[co] * rsqrtf(var[co] + BN_EPS);
      sc[i] = s;
      sh[i] = beta[co] - mean[co] * s;
    }
#pragma unroll
    for (int bn = 0; bn < 2; ++bn) {
      v8f acc = am ? (bn ? acc11 : acc10) : (bn ? acc01 : acc00);
      int p = p0 + wn * 32 + bn * 16 + (lane & 15);
      int img = p / HW, rem = p - img * HW;
      int y = rem / D, x = rem - y * D;
#pragma unroll
      for (int i = 0; i < 8; ++i) {
        float v = acc[i] * sc[i] + sh[i];
        v = v > 0.f ? v : 0.f;
        out[(((size_t)img * Cout + (mrow + i)) * D + y) * D + x] = (__bf16)v;
      }
    }
  }
}

// ---------------------------------------------------------------------------
// Generic GEMM: out[n][m] = act(A[m][:] . Bm[n][:] + bias[m])
// A: bf16 MxK row-major. Bm: bf16 NxK row-major. K multiple of 32.
// 128 threads = 4 waves; block tile 128x16; wave tile 32x16 -> 2 WMMAs/iter.
// Double-buffered LDS with b128/b64 staging for full streaming bandwidth.
// ---------------------------------------------------------------------------
template<int RELU, int STORE_BF16>
__global__ __launch_bounds__(128)
void gemm_bias_wmma(const __bf16* __restrict__ A, const __bf16* __restrict__ Bm,
                    const float* __restrict__ bias, void* __restrict__ outv,
                    int M, int K, int Mstride)
{
  __shared__ unsigned int As32[2][128 * 20];
  __shared__ unsigned int Bs32[2][16 * 20];
  const int ldw = 20;
  const int tid  = threadIdx.x;
  const int wave = tid >> 5;
  const int m0 = blockIdx.x * 128;
  const int n0 = blockIdx.y * 16;
  const int Kw = K >> 1;
  const unsigned int* A32 = (const unsigned int*)A;
  const unsigned int* B32 = (const unsigned int*)Bm;

  // A staging: thread owns full row (16 dwords); B: 2 dwords
  const int am_ = m0 + tid;
  const bool aok = am_ < M;
  const unsigned int* arow = A32 + (size_t)am_ * Kw;
  const int bn_ = tid >> 3;
  const int bc_ = (tid & 7) * 2;
  const unsigned int* brow = B32 + (size_t)(n0 + bn_) * Kw + bc_;

  v4u t0, t1, t2, t3;
  v2u tb;
  auto fetch = [&](int k0) {
    int k02 = k0 >> 1;
    v4u z = {};
    t0 = z; t1 = z; t2 = z; t3 = z;
    if (aok) {
      t0 = *(const v4u*)(arow + k02);
      t1 = *(const v4u*)(arow + k02 + 4);
      t2 = *(const v4u*)(arow + k02 + 8);
      t3 = *(const v4u*)(arow + k02 + 12);
    }
    tb = *(const v2u*)(brow + k02);
  };

  v8f acc0 = {}, acc1 = {};
  const int nIter = K >> 5;

  fetch(0);
  for (int it = 0; it < nIter; ++it) {
    unsigned int* Ab = As32[it & 1];
    unsigned int* Bb = Bs32[it & 1];
    *(v4u*)(&Ab[tid * ldw])      = t0;
    *(v4u*)(&Ab[tid * ldw + 4])  = t1;
    *(v4u*)(&Ab[tid * ldw + 8])  = t2;
    *(v4u*)(&Ab[tid * ldw + 12]) = t3;
    *(v2u*)(&Bb[bn_ * ldw + bc_]) = tb;
    __syncthreads();
    if (it + 1 < nIter) fetch((it + 1) * 32);
    v16bf a0 = fragA32(Ab + (wave * 32) * ldw, ldw);
    v16bf a1 = fragA32(Ab + (wave * 32 + 16) * ldw, ldw);
    v16bf bf = fragB32(Bb, ldw);
    acc0 = __builtin_amdgcn_wmma_f32_16x16x32_bf16(false, a0, false, bf, (short)0, acc0, false, false);
    acc1 = __builtin_amdgcn_wmma_f32_16x16x32_bf16(false, a1, false, bf, (short)0, acc1, false, false);
  }

  int lane = tid & 31;
  int n = n0 + (lane & 15);
#pragma unroll
  for (int am = 0; am < 2; ++am) {
    v8f acc = am ? acc1 : acc0;
#pragma unroll
    for (int i = 0; i < 8; ++i) {
      int m = m0 + wave * 32 + am * 16 + ((lane >> 4) << 3) + i;
      if (m < M) {
        float v = acc[i] + (bias ? bias[m] : 0.f);
        if (RELU) v = v > 0.f ? v : 0.f;
        if (STORE_BF16) ((__bf16*)outv)[(size_t)n * Mstride + m] = (__bf16)v;
        else            ((float*)outv)[(size_t)n * Mstride + m] = v;
      }
    }
  }
}

// ---------------------------------------------------------------------------
// Small helper kernels
// ---------------------------------------------------------------------------
__global__ void cvt_f32_bf16(const float* __restrict__ in, __bf16* __restrict__ out, long n) {
  long i = (long)blockIdx.x * blockDim.x + threadIdx.x;
  if (i < n) out[i] = (__bf16)in[i];
}

// convert with row padding: out[r][0..Kpad) = in[r][0..Kc) padded with zeros
__global__ void cvt_pad_bf16(const float* __restrict__ in, __bf16* __restrict__ out,
                             int Kc, int Kpad, long rows) {
  long i = (long)blockIdx.x * blockDim.x + threadIdx.x;
  if (i >= rows * (long)Kpad) return;
  long r = i / Kpad; int k = (int)(i - r * Kpad);
  out[i] = (k < Kc) ? (__bf16)in[r * Kc + k] : (__bf16)0.0f;
}

__global__ void maxpool2x2(const __bf16* __restrict__ in, __bf16* __restrict__ out,
                           int C, int D, int ntot) {
  int i = blockIdx.x * blockDim.x + threadIdx.x;
  if (i >= ntot) return;
  int od = D >> 1;
  int x = i % od; int t = i / od;
  int y = t % od; t /= od;
  int c = t % C;  int n = t / C;
  const __bf16* p = in + ((((size_t)n * C + c) * D + 2 * y) * D + 2 * x);
  float a = (float)p[0], b = (float)p[1], e = (float)p[D], d = (float)p[D + 1];
  out[i] = (__bf16)fmaxf(fmaxf(a, b), fmaxf(e, d));
}

__global__ void build_wcat(const float* __restrict__ wih, const float* __restrict__ whh,
                           __bf16* __restrict__ wcat) {
  int i = blockIdx.x * blockDim.x + threadIdx.x;
  if (i >= 2048 * 1024) return;
  int m = i >> 10, k = i & 1023;
  float v = (k < 512) ? wih[m * 512 + k] : whh[m * 512 + (k - 512)];
  wcat[i] = (__bf16)v;
}

__global__ void build_bcat(const float* __restrict__ bih, const float* __restrict__ bhh,
                           float* __restrict__ bcat) {
  int i = blockIdx.x * blockDim.x + threadIdx.x;
  if (i < 2048) bcat[i] = bih[i] + bhh[i];
}

__global__ void build_feat(const float* __restrict__ fc3o, __bf16* __restrict__ xseq) {
  int i = blockIdx.x * blockDim.x + threadIdx.x;
  if (i >= 16 * 512) return;
  int b = i >> 9, e = i & 511;
  float s = 0.f;
#pragma unroll
  for (int r = 0; r < 8; ++r) s += fc3o[b * 4096 + r * 512 + e];
  xseq[(size_t)b * 512 + e] = (__bf16)(s * 0.125f);   // t = 0 row
}

__global__ void build_embed(const float* __restrict__ emb, const int* __restrict__ cap,
                            __bf16* __restrict__ xseq) {
  int i = blockIdx.x * blockDim.x + threadIdx.x;
  if (i >= 19 * 16 * 512) return;
  int e = i & 511; int t2 = i >> 9; int b = t2 & 15; int t = (t2 >> 4) + 1;
  int tok = cap[b * 20 + t];
  xseq[((size_t)t * 16 + b) * 512 + e] = (__bf16)emb[(size_t)tok * 512 + e];
}

__global__ void init_lstm(float* __restrict__ cbuf, __bf16* __restrict__ xh) {
  int i = blockIdx.x * blockDim.x + threadIdx.x;
  if (i >= 16 * 512) return;
  int b = i >> 9, j = i & 511;
  cbuf[i] = 0.f;
  xh[b * 1024 + 512 + j] = (__bf16)0.0f;
}

__global__ void set_xt(const __bf16* __restrict__ xseq, __bf16* __restrict__ xh, int t) {
  int i = blockIdx.x * blockDim.x + threadIdx.x;
  if (i >= 16 * 512) return;
  int b = i >> 9, j = i & 511;
  xh[b * 1024 + j] = xseq[((size_t)t * 16 + b) * 512 + j];
}

__global__ void lstm_pointwise(const float* __restrict__ g, float* __restrict__ cbuf,
                               __bf16* __restrict__ xh, __bf16* __restrict__ hseq, int t) {
  int i = blockIdx.x * blockDim.x + threadIdx.x;
  if (i >= 16 * 512) return;
  int b = i >> 9, j = i & 511;
  const float* gr = g + b * 2048;
  float ig = 1.f / (1.f + __expf(-gr[j]));
  float fg = 1.f / (1.f + __expf(-gr[512 + j]));
  float gg = tanhf(gr[1024 + j]);
  float og = 1.f / (1.f + __expf(-gr[1536 + j]));
  float c = fg * cbuf[i] + ig * gg;
  cbuf[i] = c;
  float h = og * tanhf(c);
  xh[b * 1024 + 512 + j]               = (__bf16)h;
  hseq[((size_t)t * 16 + b) * 512 + j] = (__bf16)h;
}

__global__ __launch_bounds__(256)
void rowmax_argmax(const float* __restrict__ logits, int ld, int V,
                   float* __restrict__ vals, int* __restrict__ idxs) {
  __shared__ float sv[256];
  __shared__ int   si[256];
  int row = blockIdx.x;
  const float* r = logits + (size_t)row * ld;
  float best = -INFINITY; int bi = 0;
  for (int j = threadIdx.x; j < V; j += 256) {
    float v = r[j];
    if (v > best) { best = v; bi = j; }
  }
  sv[threadIdx.x] = best; si[threadIdx.x] = bi;
  __syncthreads();
  for (int s = 128; s > 0; s >>= 1) {
    if (threadIdx.x < s) {
      float ov = sv[threadIdx.x + s]; int oi = si[threadIdx.x + s];
      if (ov > sv[threadIdx.x] || (ov == sv[threadIdx.x] && oi < si[threadIdx.x])) {
        sv[threadIdx.x] = ov; si[threadIdx.x] = oi;
      }
    }
    __syncthreads();
  }
  if (threadIdx.x == 0) { vals[row] = sv[0]; idxs[row] = si[0]; }
}

// ---------------------------------------------------------------------------
// Host orchestration
// ---------------------------------------------------------------------------
extern "C" void kernel_launch(void* const* d_in, const int* in_sizes, int n_in,
                              void* d_out, int out_size, void* d_ws, size_t ws_size,
                              hipStream_t stream) {
  (void)in_sizes; (void)n_in; (void)out_size; (void)ws_size;
  const int V = 10000, T = 20, VP = 10048;

  struct L { int cin, cout, dim; };
  const L convs[13] = {{3,64,224},{64,64,224},{64,128,112},{128,128,112},
                       {128,256,56},{256,256,56},{256,256,56},
                       {256,512,28},{512,512,28},{512,512,28},
                       {512,512,14},{512,512,14},{512,512,14}};
  const int pool_after[13] = {0,1,0,1,0,0,1,0,0,1,0,0,1};

  char* ws = (char*)d_ws; size_t off = 0;
  auto carve = [&](size_t bytes) -> void* {
    void* p = ws + off; off += (bytes + 255) & ~(size_t)255; return p;
  };

  __bf16* actA = (__bf16*)carve((size_t)64 * 224 * 224 * 16 * 2);
  __bf16* actB = (__bf16*)carve((size_t)64 * 224 * 224 * 16 * 2);
  size_t cwe[13]; size_t cwtot = 0;
  int kpadv[13];
  for (int l = 0; l < 13; ++l) {
    int kc = convs[l].cin * 9;
    kpadv[l] = (kc + 31) & ~31;
    cwe[l] = (size_t)convs[l].cout * kpadv[l];
    cwtot += cwe[l];
  }
  __bf16* convw = (__bf16*)carve(cwtot * 2);
  __bf16* fc1w  = (__bf16*)carve((size_t)4096 * 25088 * 2);
  __bf16* fc2w  = (__bf16*)carve((size_t)4096 * 4096 * 2);
  __bf16* fc3w  = (__bf16*)carve((size_t)4096 * 4096 * 2);
  __bf16* wcat  = (__bf16*)carve((size_t)2048 * 1024 * 2);
  __bf16* fcWb  = (__bf16*)carve((size_t)V * 512 * 2);
  __bf16* fc1o  = (__bf16*)carve((size_t)16 * 4096 * 2);
  __bf16* fc2o  = (__bf16*)carve((size_t)16 * 4096 * 2);
  float*  fc3o  = (float*) carve((size_t)16 * 4096 * 4);
  __bf16* xseq  = (__bf16*)carve((size_t)T * 16 * 512 * 2);
  __bf16* hseq  = (__bf16*)carve((size_t)T * 16 * 512 * 2);
  __bf16* xh    = (__bf16*)carve((size_t)16 * 1024 * 2);
  float*  cbuf  = (float*) carve((size_t)16 * 512 * 4);
  float*  gbuf  = (float*) carve((size_t)16 * 2048 * 4);
  float*  bcat  = (float*) carve((size_t)2048 * 4);
  float*  logits= (float*) carve((size_t)320 * VP * 4);

  auto cvt = [&](const void* src, __bf16* dst, long n) {
    cvt_f32_bf16<<<(unsigned)((n + 255) / 256), 256, 0, stream>>>((const float*)src, dst, n);
  };

  // --- weight / input conversions to bf16
  cvt(d_in[0], actA, (long)16 * 3 * 224 * 224);
  {
    size_t w = 0;
    for (int l = 0; l < 13; ++l) {
      int kc = convs[l].cin * 9;
      long n = (long)convs[l].cout * kpadv[l];
      cvt_pad_bf16<<<(unsigned)((n + 255) / 256), 256, 0, stream>>>(
          (const float*)d_in[2 + 5 * l], convw + w, kc, kpadv[l], convs[l].cout);
      w += cwe[l];
    }
  }
  cvt(d_in[67], fc1w, (long)4096 * 25088);
  cvt(d_in[69], fc2w, (long)4096 * 4096);
  cvt(d_in[71], fc3w, (long)4096 * 4096);
  cvt(d_in[78], fcWb, (long)V * 512);
  build_wcat<<<(2048 * 1024 + 255) / 256, 256, 0, stream>>>((const float*)d_in[74], (const float*)d_in[75], wcat);
  build_bcat<<<8, 256, 0, stream>>>((const float*)d_in[76], (const float*)d_in[77], bcat);

  // --- VGG conv stack (ping-pong bf16 activations)
  __bf16* cur = actA; __bf16* nxt = actB;
  size_t woff = 0;
  for (int l = 0; l < 13; ++l) {
    const L& c = convs[l];
    dim3 grid(c.cout / 64, (16 * c.dim * c.dim) / 64);
    conv3x3_bn_relu_wmma<<<grid, 128, 0, stream>>>(
        cur, convw + woff,
        (const float*)d_in[3 + 5 * l], (const float*)d_in[4 + 5 * l],
        (const float*)d_in[5 + 5 * l], (const float*)d_in[6 + 5 * l],
        nxt, c.cin, c.cout, c.dim);
    woff += cwe[l];
    { __bf16* t = cur; cur = nxt; nxt = t; }
    if (pool_after[l]) {
      int od = c.dim / 2;
      int ntot = 16 * c.cout * od * od;
      maxpool2x2<<<(ntot + 255) / 256, 256, 0, stream>>>(cur, nxt, c.cout, c.dim, ntot);
      { __bf16* t = cur; cur = nxt; nxt = t; }
    }
  }
  // cur now holds (16,512,7,7) bf16 == row-major (16, 25088)

  // --- VGG classifier head
  gemm_bias_wmma<1, 1><<<dim3(4096 / 128, 1), 128, 0, stream>>>(fc1w, cur,  (const float*)d_in[68], fc1o, 4096, 25088, 4096);
  gemm_bias_wmma<1, 1><<<dim3(4096 / 128, 1), 128, 0, stream>>>(fc2w, fc1o, (const float*)d_in[70], fc2o, 4096, 4096, 4096);
  gemm_bias_wmma<0, 0><<<dim3(4096 / 128, 1), 128, 0, stream>>>(fc3w, fc2o, (const float*)d_in[72], fc3o, 4096, 4096, 4096);

  // --- LSTM input sequence
  build_feat<<<(16 * 512 + 255) / 256, 256, 0, stream>>>(fc3o, xseq);
  build_embed<<<(19 * 16 * 512 + 255) / 256, 256, 0, stream>>>((const float*)d_in[73], (const int*)d_in[1], xseq);
  init_lstm<<<(16 * 512 + 255) / 256, 256, 0, stream>>>(cbuf, xh);

  // --- LSTM timesteps: gates = [x,h] @ Wcat^T + bcat, then pointwise
  for (int t = 0; t < T; ++t) {
    set_xt<<<(16 * 512 + 255) / 256, 256, 0, stream>>>(xseq, xh, t);
    gemm_bias_wmma<0, 0><<<dim3(2048 / 128, 1), 128, 0, stream>>>(wcat, xh, bcat, gbuf, 2048, 1024, 2048);
    lstm_pointwise<<<(16 * 512 + 255) / 256, 256, 0, stream>>>(gbuf, cbuf, xh, hseq, t);
  }

  // --- logits: hs (320 x 512) @ fc_W^T (10000 x 512) + fc_b
  gemm_bias_wmma<0, 0><<<dim3((V + 127) / 128, 320 / 16), 128, 0, stream>>>(fcWb, hseq, (const float*)d_in[79], logits, V, 512, VP);

  // --- per-(t,b) max + argmax over vocab
  float* vals = (float*)d_out;
  int*   idxs = (int*)(vals + 320);
  rowmax_argmax<<<320, 256, 0, stream>>>(logits, VP, V, vals, idxs);
}
